// MultiHeadAttention_73890617361024
// MI455X (gfx1250) — matrix operand order repaired and checked by reading, not compiled, not run on hardware
//
#include <hip/hip_runtime.h>

typedef __attribute__((ext_vector_type(16))) __bf16 v16bf;
typedef __attribute__((ext_vector_type(8)))  float  v8f;

__device__ __forceinline__ unsigned short f32_to_bf16(float f) {
  unsigned int u = __builtin_bit_cast(unsigned int, f);
  u += 0x7FFFu + ((u >> 16) & 1u);  // round-to-nearest-even
  return (unsigned short)(u >> 16);
}
__device__ __forceinline__ unsigned long long pack4_bf16(float a, float b, float c,
                                                         float d) {
  return (unsigned long long)f32_to_bf16(a) |
         ((unsigned long long)f32_to_bf16(b) << 16) |
         ((unsigned long long)f32_to_bf16(c) << 32) |
         ((unsigned long long)f32_to_bf16(d) << 48);
}
__device__ __forceinline__ unsigned long long pack4_us(unsigned short a,
                                                       unsigned short b,
                                                       unsigned short c,
                                                       unsigned short d) {
  return (unsigned long long)a | ((unsigned long long)b << 16) |
         ((unsigned long long)c << 32) | ((unsigned long long)d << 48);
}

// Two 16B LDS loads -> one 16-element bf16 fragment.
__device__ __forceinline__ v16bf load_frag16(const unsigned short* p0,
                                             const unsigned short* p1) {
  union { uint4 u[2]; v16bf v; } f;
  f.u[0] = *(const uint4*)p0;
  f.u[1] = *(const uint4*)p1;
  return f.v;
}

// A fragment (16x32): row-major tile, element (m,k) at tile[m*stride + k].
// Lane layout: m=lane&15; elems 0..7 = k kb..kb+7, elems 8..15 = k kb+16..kb+23,
// kb = (lane&16)?8:0  -> two contiguous b128 loads.
__device__ __forceinline__ v16bf load_a_frag(const unsigned short* tile, int stride) {
  const int lane = threadIdx.x & 31;
  const unsigned short* p = tile + (lane & 15) * stride + ((lane & 16) ? 8 : 0);
  return load_frag16(p, p + 16);
}

// B fragment (32x16): k-contiguous storage, element (k,n) at tileT[n*stride + k].
// Lane layout: n=lane&15; elems 0..7 = k kb..kb+7, elems 8..15 = k kb+16..kb+23,
// kb = (lane&16)?8:0 -> two b128 loads.
__device__ __forceinline__ v16bf load_b_frag(const unsigned short* tileT, int stride) {
  const int lane = threadIdx.x & 31;
  const unsigned short* p = tileT + (lane & 15) * stride + ((lane & 16) ? 8 : 0);
  return load_frag16(p, p + 16);
}

__device__ __forceinline__ v8f wmma_bf16(v16bf a, v16bf b, v8f c) {
  return __builtin_amdgcn_wmma_f32_16x16x32_bf16(false, a, false, b, (short)0, c,
                                                 false, false);
}

// ---------------------------------------------------------------------------
// Kernel 1: qkv = x @ W_qkv + b_qkv -> Q/K/V [B,H,S,D] bf16
// M=4096, K=2048, N=6144. 128x128 tile, 256 thr (8 waves), BK=64.
// ---------------------------------------------------------------------------
__global__ __launch_bounds__(256) void qkv_gemm_kernel(
    const float* __restrict__ x, const float* __restrict__ Wqkv,
    const float* __restrict__ bqkv, unsigned short* __restrict__ qws,
    unsigned short* __restrict__ kws, unsigned short* __restrict__ vws) {
  constexpr int Kdim = 2048, Nqkv = 6144;
  __shared__ alignas(16) unsigned short As[128 * 72];   // [m][k], k-major
  __shared__ alignas(16) unsigned short Bt[128 * 72];   // [n][k], transposed
  const int tid = threadIdx.x;
  const int row0 = blockIdx.y * 128;
  const int col0 = blockIdx.x * 128;
  const int wid = tid >> 5;
  const int wave_m = wid >> 1;  // 0..3, 32 rows
  const int wave_n = wid & 1;   // 0..1, 64 cols

  v8f acc[2][4];
#pragma unroll
  for (int i = 0; i < 2; ++i)
#pragma unroll
    for (int j = 0; j < 4; ++j)
#pragma unroll
      for (int r = 0; r < 8; ++r) acc[i][j][r] = 0.0f;

  const int an = tid & 127;       // B-load: n column
  const int akg = tid >> 7;       // B-load: k group (0..1)

  for (int k0 = 0; k0 < Kdim; k0 += 64) {
    __syncthreads();
    // A tile: 128x64 fp32 -> bf16, float4 loads, b64 LDS stores
#pragma unroll
    for (int i = 0; i < 8; ++i) {
      int q = i * 256 + tid;
      int r = q >> 4, c4 = (q & 15) * 4;
      float4 f = *(const float4*)&x[(size_t)(row0 + r) * Kdim + k0 + c4];
      *(unsigned long long*)&As[r * 72 + c4] = pack4_bf16(f.x, f.y, f.z, f.w);
    }
    // B tile: 64x128 fp32 -> bf16 transposed into Bt[n][k]
#pragma unroll
    for (int i = 0; i < 8; ++i) {
      int k4 = (i * 2 + akg) * 4;
      float w0 = Wqkv[(size_t)(k0 + k4 + 0) * Nqkv + col0 + an];
      float w1 = Wqkv[(size_t)(k0 + k4 + 1) * Nqkv + col0 + an];
      float w2 = Wqkv[(size_t)(k0 + k4 + 2) * Nqkv + col0 + an];
      float w3 = Wqkv[(size_t)(k0 + k4 + 3) * Nqkv + col0 + an];
      *(unsigned long long*)&Bt[an * 72 + k4] = pack4_bf16(w0, w1, w2, w3);
    }
    __syncthreads();
#pragma unroll
    for (int ks = 0; ks < 2; ++ks) {
      v16bf a[2], b[4];
#pragma unroll
      for (int mt = 0; mt < 2; ++mt)
        a[mt] = load_a_frag(&As[(wave_m * 32 + mt * 16) * 72 + ks * 32], 72);
#pragma unroll
      for (int nt = 0; nt < 4; ++nt)
        b[nt] = load_b_frag(&Bt[(wave_n * 64 + nt * 16) * 72 + ks * 32], 72);
#pragma unroll
      for (int mt = 0; mt < 2; ++mt)
#pragma unroll
        for (int nt = 0; nt < 4; ++nt)
          acc[mt][nt] = wmma_bf16(a[mt], b[nt], acc[mt][nt]);
    }
  }

  const int lane = tid & 31;
  const int mofs = (lane & 16) ? 8 : 0;
  const int nofs = lane & 15;
#pragma unroll
  for (int mt = 0; mt < 2; ++mt)
#pragma unroll
    for (int nt = 0; nt < 4; ++nt) {
      int n = col0 + wave_n * 64 + nt * 16 + nofs;
      float bias = bqkv[n];
      int sec = n >> 11;  // 0=Q, 1=K, 2=V (wave-uniform per n-tile)
      int wi = n & 2047;
      int h = wi >> 7, d = wi & 127;
      // branchless destination select
      unsigned short* dstbuf = (sec == 0) ? qws : ((sec == 1) ? kws : vws);
#pragma unroll
      for (int r = 0; r < 8; ++r) {
        int m = row0 + wave_m * 32 + mt * 16 + r + mofs;
        int bb = m >> 11, s = m & 2047;
        size_t dst = (((size_t)bb * 16 + h) * 2048 + s) * 128 + d;
        dstbuf[dst] = f32_to_bf16(acc[mt][nt][r] + bias);
      }
    }
}

// ---------------------------------------------------------------------------
// Kernel 2: flash attention per (b,h). 64 Q rows/block, 4 waves x 16 rows.
// ---------------------------------------------------------------------------
__global__ __launch_bounds__(128) void flash_attn_kernel(
    const unsigned short* __restrict__ qws, const unsigned short* __restrict__ kws,
    const unsigned short* __restrict__ vws, unsigned short* __restrict__ aws) {
  constexpr int S = 2048, D = 128;
  __shared__ alignas(16) unsigned short Qs[64 * 136];   // [q][d] row-major
  __shared__ alignas(16) unsigned short Ks[32 * 136];   // [kv][d] (k-contig for QK^T)
  __shared__ alignas(16) unsigned short Vt[128 * 40];   // [d][kv] (k-contig for PV)
  __shared__ alignas(16) unsigned short Ps[4 * 16 * 40];
  const int tid = threadIdx.x;
  const int w = tid >> 5;
  const int lane = tid & 31;
  const int q0 = blockIdx.x * 64;
  const int bh = blockIdx.y;
  const size_t base = (size_t)bh * S * D;

  // Q tile: 64x128 bf16, u64 copies
#pragma unroll
  for (int i = 0; i < 8; ++i) {
    int q = i * 128 + tid;
    int r = q >> 4, c8 = (q & 15) * 8;
    *(unsigned long long*)&Qs[r * 136 + c8] =
        *(const unsigned long long*)&qws[base + (size_t)(q0 + r) * D + c8];
  }

  float mrow[8], lrow[8], corr[8];
  v8f o[8];
#pragma unroll
  for (int r = 0; r < 8; ++r) { mrow[r] = -1e30f; lrow[r] = 0.0f; }
#pragma unroll
  for (int dt = 0; dt < 8; ++dt)
#pragma unroll
    for (int r = 0; r < 8; ++r) o[dt][r] = 0.0f;

  const float scale = 0.088388347648318447f;  // 1/sqrt(128)
  unsigned short* Pw = &Ps[w * 16 * 40];
  const int mofs = (lane & 16) ? 8 : 0;
  const int nofs = lane & 15;

  for (int kv0 = 0; kv0 < S; kv0 += 32) {
    __syncthreads();
    // K tile: straight u64 copy (row-major)
#pragma unroll
    for (int i = 0; i < 4; ++i) {
      int q = i * 128 + tid;
      int r = q >> 4, c8 = (q & 15) * 8;
      *(unsigned long long*)&Ks[r * 136 + c8] =
          *(const unsigned long long*)&kws[base + (size_t)(kv0 + r) * D + c8];
    }
    // V tile: transpose into Vt[d][kv]; gather 4 kv per thread, b64 store
#pragma unroll
    for (int i = 0; i < 8; ++i) {
      int k4 = i * 4;
      unsigned short v0 = vws[base + (size_t)(kv0 + k4 + 0) * D + tid];
      unsigned short v1 = vws[base + (size_t)(kv0 + k4 + 1) * D + tid];
      unsigned short v2 = vws[base + (size_t)(kv0 + k4 + 2) * D + tid];
      unsigned short v3 = vws[base + (size_t)(kv0 + k4 + 3) * D + tid];
      *(unsigned long long*)&Vt[tid * 40 + k4] = pack4_us(v0, v1, v2, v3);
    }
    __syncthreads();

    // scores: S[16x32] = Q * K^T  (B element (k=d,n=kv) = Ks[n*136 + k])
    v8f s0, s1;
#pragma unroll
    for (int r = 0; r < 8; ++r) { s0[r] = 0.0f; s1[r] = 0.0f; }
#pragma unroll
    for (int ks = 0; ks < 4; ++ks) {
      v16bf a = load_a_frag(&Qs[(w * 16) * 136 + ks * 32], 136);
      v16bf b0 = load_b_frag(&Ks[0 * 136 + ks * 32], 136);
      v16bf b1 = load_b_frag(&Ks[16 * 136 + ks * 32], 136);
      s0 = wmma_bf16(a, b0, s0);
      s1 = wmma_bf16(a, b1, s1);
    }

    // online softmax: row m=r+mofs lives in one 16-lane half
#pragma unroll
    for (int r = 0; r < 8; ++r) {
      float v0 = s0[r] * scale;
      float v1 = s1[r] * scale;
      float rmax = fmaxf(v0, v1);
#pragma unroll
      for (int off = 1; off < 16; off <<= 1)
        rmax = fmaxf(rmax, __shfl_xor(rmax, off, 32));
      float mnew = fmaxf(mrow[r], rmax);
      float c = __expf(mrow[r] - mnew);
      float p0 = __expf(v0 - mnew);
      float p1 = __expf(v1 - mnew);
      float rsum = p0 + p1;
#pragma unroll
      for (int off = 1; off < 16; off <<= 1)
        rsum += __shfl_xor(rsum, off, 32);
      lrow[r] = lrow[r] * c + rsum;
      mrow[r] = mnew;
      corr[r] = c;
      int m = r + mofs;
      Pw[m * 40 + nofs]      = f32_to_bf16(p0);
      Pw[m * 40 + 16 + nofs] = f32_to_bf16(p1);
    }
#pragma unroll
    for (int dt = 0; dt < 8; ++dt)
#pragma unroll
      for (int r = 0; r < 8; ++r) o[dt][r] *= corr[r];

    asm volatile("s_wait_dscnt 0" ::: "memory");  // own-wave P visibility

    // O[16x128] += P[16x32] * V[32x128]   (B element (k=kv,n=d) = Vt[n*40 + k])
    v16bf pa = load_a_frag(Pw, 40);
#pragma unroll
    for (int dt = 0; dt < 8; ++dt) {
      v16bf bv = load_b_frag(&Vt[(dt * 16) * 40], 40);
      o[dt] = wmma_bf16(pa, bv, o[dt]);
    }
  }

  // normalize + write [B,S,E] bf16 (merged heads)
  const int b = bh >> 4, h = bh & 15;
#pragma unroll
  for (int r = 0; r < 8; ++r) {
    float inv = 1.0f / lrow[r];
    int s = q0 + w * 16 + r + mofs;
#pragma unroll
    for (int dt = 0; dt < 8; ++dt) {
      int col = h * 128 + dt * 16 + nofs;
      aws[((size_t)b * 2048 + s) * 2048 + col] = f32_to_bf16(o[dt][r] * inv);
    }
  }
}

// ---------------------------------------------------------------------------
// Kernel 3: out = attn @ W_proj + b_proj. M=4096, N=2048, K=2048, fp32 out.
// ---------------------------------------------------------------------------
__global__ __launch_bounds__(256) void proj_gemm_kernel(
    const unsigned short* __restrict__ aws, const float* __restrict__ Wp,
    const float* __restrict__ bp, float* __restrict__ out) {
  constexpr int Kdim = 2048, N = 2048;
  __shared__ alignas(16) unsigned short As[128 * 72];
  __shared__ alignas(16) unsigned short Bt[128 * 72];   // [n][k] transposed
  const int tid = threadIdx.x;
  const int row0 = blockIdx.y * 128;
  const int col0 = blockIdx.x * 128;
  const int wid = tid >> 5;
  const int wave_m = wid >> 1;
  const int wave_n = wid & 1;

  v8f acc[2][4];
#pragma unroll
  for (int i = 0; i < 2; ++i)
#pragma unroll
    for (int j = 0; j < 4; ++j)
#pragma unroll
      for (int r = 0; r < 8; ++r) acc[i][j][r] = 0.0f;

  const int an = tid & 127;
  const int akg = tid >> 7;

  for (int k0 = 0; k0 < Kdim; k0 += 64) {
    __syncthreads();
    // A tile: bf16 ws, straight u64 copy
#pragma unroll
    for (int i = 0; i < 4; ++i) {
      int q = i * 256 + tid;
      int r = q >> 3, c8 = (q & 7) * 8;
      *(unsigned long long*)&As[r * 72 + c8] =
          *(const unsigned long long*)&aws[(size_t)(row0 + r) * Kdim + k0 + c8];
    }
    // B tile: fp32 -> bf16, transposed into Bt[n][k]
#pragma unroll
    for (int i = 0; i < 8; ++i) {
      int k4 = (i * 2 + akg) * 4;
      float w0 = Wp[(size_t)(k0 + k4 + 0) * N + col0 + an];
      float w1 = Wp[(size_t)(k0 + k4 + 1) * N + col0 + an];
      float w2 = Wp[(size_t)(k0 + k4 + 2) * N + col0 + an];
      float w3 = Wp[(size_t)(k0 + k4 + 3) * N + col0 + an];
      *(unsigned long long*)&Bt[an * 72 + k4] = pack4_bf16(w0, w1, w2, w3);
    }
    __syncthreads();
#pragma unroll
    for (int ks = 0; ks < 2; ++ks) {
      v16bf a[2], b[4];
#pragma unroll
      for (int mt = 0; mt < 2; ++mt)
        a[mt] = load_a_frag(&As[(wave_m * 32 + mt * 16) * 72 + ks * 32], 72);
#pragma unroll
      for (int nt = 0; nt < 4; ++nt)
        b[nt] = load_b_frag(&Bt[(wave_n * 64 + nt * 16) * 72 + ks * 32], 72);
#pragma unroll
      for (int mt = 0; mt < 2; ++mt)
#pragma unroll
        for (int nt = 0; nt < 4; ++nt)
          acc[mt][nt] = wmma_bf16(a[mt], b[nt], acc[mt][nt]);
    }
  }

  const int lane = tid & 31;
  const int mofs = (lane & 16) ? 8 : 0;
  const int nofs = lane & 15;
#pragma unroll
  for (int mt = 0; mt < 2; ++mt)
#pragma unroll
    for (int nt = 0; nt < 4; ++nt) {
      int n = col0 + wave_n * 64 + nt * 16 + nofs;
      float bias = bp[n];
#pragma unroll
      for (int r = 0; r < 8; ++r) {
        int m = row0 + wave_m * 32 + mt * 16 + r + mofs;
        out[(size_t)m * N + n] = acc[mt][nt][r] + bias;
      }
    }
}

// ---------------------------------------------------------------------------
extern "C" void kernel_launch(void* const* d_in, const int* in_sizes, int n_in,
                              void* d_out, int out_size, void* d_ws, size_t ws_size,
                              hipStream_t stream) {
  const float* x     = (const float*)d_in[0];
  const float* Wqkv  = (const float*)d_in[1];
  const float* bqkv  = (const float*)d_in[2];
  const float* Wproj = (const float*)d_in[3];
  const float* bproj = (const float*)d_in[4];
  float* out = (float*)d_out;

  const size_t qkv_elems = (size_t)2 * 16 * 2048 * 128;
  unsigned short* qws = (unsigned short*)d_ws;
  unsigned short* kws = qws + qkv_elems;
  unsigned short* vws = kws + qkv_elems;
  unsigned short* aws = vws + qkv_elems;

  qkv_gemm_kernel<<<dim3(48, 32), 256, 0, stream>>>(x, Wqkv, bqkv, qws, kws, vws);
  flash_attn_kernel<<<dim3(32, 32), 128, 0, stream>>>(qws, kws, vws, aws);
  proj_gemm_kernel<<<dim3(16, 32), 256, 0, stream>>>(aws, Wproj, bproj, out);
}
